// TopicFormer_52922587022041
// MI455X (gfx1250) — matrix operand order repaired and checked by reading, not compiled, not run on hardware
//
#include <hip/hip_runtime.h>
#include <hip/hip_bf16.h>
#include <math.h>

// ---------------- CDNA5 WMMA types ----------------
typedef __bf16 v16bf __attribute__((ext_vector_type(16)));
typedef __bf16 v8bf  __attribute__((ext_vector_type(8)));
typedef float  v8f   __attribute__((ext_vector_type(8)));

#define D_MODEL 256
#define NHEAD   8
#define HDIM    32
#define NTOP    100
#define NSAMP   6
#define NITER   2
#define NB      2
#define LL      4800
#define LTOT    9600
#define EPS_    1e-6f

// pack two f32 -> packed bf16 pair (round-to-nearest-even)
__device__ inline unsigned pk2(float a, float b){
    union { float f; unsigned u; } x, y; x.f = a; y.f = b;
    unsigned ra = (x.u + 0x7FFFu + ((x.u >> 16) & 1u)) >> 16;
    unsigned rb = (y.u + 0x7FFFu + ((y.u >> 16) & 1u)) >> 16;
    return ra | (rb << 16);
}

__device__ inline float wave_sum(float v){ for(int o=16;o>0;o>>=1) v += __shfl_xor(v,o,32); return v; }
__device__ inline float wave_max(float v){ for(int o=16;o>0;o>>=1) v = fmaxf(v,__shfl_xor(v,o,32)); return v; }
__device__ inline int   wave_imin(int v){ for(int o=16;o>0;o>>=1){ int w=__shfl_xor(v,o,32); v = v<w?v:w; } return v; }

// ---------------- WMMA GEMM (NT): C[M,N] = alpha * A[M,K] @ B[N,K]^T ----------------
// A,B f32 row-major with inner dim K; bf16-converted into LDS tiles laid out so
// each lane's WMMA fragment is two contiguous 16B runs -> ds_load_b128 pairs.
// 128 threads = 4 waves; 64x64 C tile; BK=64 (2 WMMA k-steps per barrier).
// Batched via blockIdx.z (strides sA/sB/sC).

// stage a 64x64 f32 tile (rows rowBase.., inner offset kBase) -> bf16 LDS [64][72]
__device__ inline void stage_tile(const float* __restrict__ G, int ld, int rowBase,
                                  int nRows, int kBase, __bf16 (*L)[72], int tid)
{
    const int r  = tid >> 1;
    const int c0 = (tid & 1) * 32;
    uint4* dst = (uint4*)&L[r][c0];
    if (rowBase + r < nRows) {
        const float4* src = (const float4*)(G + (long)(rowBase + r) * ld + kBase + c0);
        #pragma unroll
        for (int t = 0; t < 4; ++t) {
            float4 a = src[2*t], b = src[2*t+1];
            dst[t] = make_uint4(pk2(a.x,a.y), pk2(a.z,a.w), pk2(b.x,b.y), pk2(b.z,b.w));
        }
    } else {
        uint4 z = make_uint4(0,0,0,0);
        #pragma unroll
        for (int t = 0; t < 4; ++t) dst[t] = z;
    }
}

__device__ inline v16bf frag2(const __bf16* p0, const __bf16* p1){
    v8bf lo = *(const v8bf*)p0;
    v8bf hi = *(const v8bf*)p1;
    return __builtin_shufflevector(lo, hi, 0,1,2,3,4,5,6,7,8,9,10,11,12,13,14,15);
}

__global__ __launch_bounds__(128)
void k_gemm(const float* __restrict__ A, const float* __restrict__ B,
            float* __restrict__ C, int M, int N, int K,
            long sA, long sB, long sC, float alpha)
{
    __shared__ __bf16 As[64][72];
    __shared__ __bf16 Bs[64][72];
    const int bz = blockIdx.z;
    A += (long)bz * sA; B += (long)bz * sB; C += (long)bz * sC;
    const int tid  = threadIdx.x;
    const int lane = tid & 31;
    const int wave = tid >> 5;
    const int row0 = blockIdx.y * 64;
    const int col0 = blockIdx.x * 64;
    const int wr = (wave >> 1) * 32;
    const int wc = (wave & 1) * 32;

    v8f acc[2][2];
    for (int i=0;i<2;i++) for(int j=0;j<2;j++) for(int r=0;r<8;r++) acc[i][j][r] = 0.f;

    const int khalf = (lane >> 4) * 8;   // A: K split 0..7 / 8..15 (+16 for hi run)
    const int kb16  = (lane >> 4) * 16;  // B: K split 0..7 / 8..15 (+16 for hi run), via kb16/2

    for (int k0 = 0; k0 < K; k0 += 64) {
        stage_tile(A, K, row0, M, k0, As, tid);
        stage_tile(B, K, col0, N, k0, Bs, tid);
        { // prefetch next K tile (gfx1250 global_prefetch_b8)
            int r = tid >> 1, c0p = (tid & 1) * 32, kn = k0 + 64;
            if (kn < K) {
                if (row0 + r < M) __builtin_prefetch(A + (long)(row0 + r)*K + kn + c0p, 0, 0);
                if (col0 + r < N) __builtin_prefetch(B + (long)(col0 + r)*K + kn + c0p, 0, 0);
            }
        }
        __syncthreads();
        #pragma unroll
        for (int kc = 0; kc < 64; kc += 32) {
            v16bf afr[2], bfr[2];
            #pragma unroll
            for (int i = 0; i < 2; ++i) {
                int ar = wr + i*16 + (lane & 15);
                afr[i] = frag2(&As[ar][kc + khalf], &As[ar][kc + khalf + 16]);
            }
            #pragma unroll
            for (int j = 0; j < 2; ++j) {
                int bc = wc + j*16 + (lane & 15);
                bfr[j] = frag2(&Bs[bc][kc + kb16/2], &Bs[bc][kc + kb16/2 + 16]);
            }
            #pragma unroll
            for (int i = 0; i < 2; ++i)
                #pragma unroll
                for (int j = 0; j < 2; ++j)
                    acc[i][j] = __builtin_amdgcn_wmma_f32_16x16x32_bf16(
                        false, afr[i], false, bfr[j], (short)0, acc[i][j], false, false);
        }
        __syncthreads();
    }
    // C frag layout: VGPR r -> M = r + 8*(lane>=16); N = lane&15
    for (int i = 0; i < 2; ++i)
        for (int j = 0; j < 2; ++j) {
            int n = col0 + wc + j*16 + (lane & 15);
            int mb = row0 + wr + i*16 + ((lane >> 4) ? 8 : 0);
            if (n < N)
                for (int r = 0; r < 8; ++r) {
                    int m = mb + r;
                    if (m < M) C[(long)m * N + n] = alpha * acc[i][j][r];
                }
        }
}

// wt[n*K+k] = w[k*N+n]  (one-time weight transpose so all GEMMs take the NT path)
__global__ void k_transpose(const float* __restrict__ w, float* __restrict__ wt, int K, int N){
    long i = blockIdx.x*(long)blockDim.x + threadIdx.x;
    if (i >= (long)K*N) return;
    int n = (int)(i / K), k = (int)(i % K);
    wt[i] = w[(long)k*N + n];
}

// ---------------- elementwise / attention helpers ----------------
__global__ void k_zero(float* p, long n){
    long i = blockIdx.x*(long)blockDim.x + threadIdx.x; if (i < n) p[i] = 0.f;
}
__global__ void k_copy(float* d, const float* s, long n){
    long i = blockIdx.x*(long)blockDim.x + threadIdx.x; if (i < n) d[i] = s[i];
}
__global__ void k_actmask(float* p, const float* mask, long n, int do_elu){
    long i = blockIdx.x*(long)blockDim.x + threadIdx.x; if (i >= n) return;
    float x = p[i];
    if (do_elu) x = (x > 0.f) ? x + 1.f : __expf(x);
    if (mask) x *= mask[i >> 8];
    p[i] = x;
}
__global__ __launch_bounds__(1024)
void k_kv(const float* __restrict__ K, const float* __restrict__ V,
          float* __restrict__ KV, float* __restrict__ Ksum, int S)
{
    __shared__ float Kc[32][33];
    __shared__ float Vc[32][33];
    int bh = blockIdx.x, b = bh >> 3, h = bh & 7;
    int tid = threadIdx.x, d = tid >> 5, v = tid & 31;
    float acc = 0.f, ks = 0.f;
    for (int s0 = 0; s0 < S; s0 += 32) {
        int sr = tid >> 5, c = tid & 31, gs = s0 + sr;
        float kv_ = 0.f, vv_ = 0.f;
        if (gs < S) {
            long base = ((long)(b*S + gs))*D_MODEL + h*HDIM + c;
            kv_ = K[base]; vv_ = V[base];
        }
        Kc[sr][c] = kv_; Vc[sr][c] = vv_;
        __syncthreads();
        for (int s = 0; s < 32; ++s) acc += Kc[s][d] * Vc[s][v];
        if (v == 0) for (int s = 0; s < 32; ++s) ks += Kc[s][d];
        __syncthreads();
    }
    KV[((long)bh*32 + d)*32 + v] = acc;
    if (v == 0) Ksum[bh*32 + d] = ks;
}
__global__ void k_attn_out(const float* __restrict__ Q, const float* __restrict__ KV,
                           const float* __restrict__ Ksum, float* __restrict__ out, int L)
{
    long w = blockIdx.x*8L + (threadIdx.x >> 5);
    int lane = threadIdx.x & 31;
    if (w >= (long)NB*L*NHEAD) return;
    int h = (int)(w % NHEAD);
    long bl = w / NHEAD;
    int b = (int)(bl / L);
    const float* qp  = Q + bl*D_MODEL + h*HDIM;
    const float* kvp = KV + ((long)(b*NHEAD + h))*1024;
    const float* ksp = Ksum + (b*NHEAD + h)*32;
    float myq = qp[lane];
    float zden = wave_sum(myq * ksp[lane]);
    float acc = 0.f;
    for (int d = 0; d < 32; ++d)
        acc += __shfl(myq, d, 32) * kvp[d*32 + lane];
    out[bl*D_MODEL + h*HDIM + lane] = acc / (zden + EPS_);
}
__global__ void k_ln(const float* in, const float* g, const float* bta, float* out, long rows){
    long w = blockIdx.x*8L + (threadIdx.x >> 5);
    if (w >= rows) return;
    int lane = threadIdx.x & 31;
    const float* r = in + w*256;
    float loc[8], s = 0.f;
    for (int t=0;t<8;t++){ loc[t] = r[lane*8 + t]; s += loc[t]; }
    s = wave_sum(s);
    float mean = s * (1.f/256.f), vs = 0.f;
    for (int t=0;t<8;t++){ float d = loc[t]-mean; vs += d*d; }
    vs = wave_sum(vs);
    float inv = rsqrtf(vs*(1.f/256.f) + 1e-5f);
    for (int t=0;t<8;t++){
        int c = lane*8+t;
        out[w*256+c] = (loc[t]-mean)*inv*g[c] + bta[c];
    }
}
__global__ void k_concat(const float* x, const float* m, float* h, long rows){
    long i = blockIdx.x*(long)blockDim.x + threadIdx.x;
    if (i >= rows*512) return;
    long rr = i >> 9; int c = (int)(i & 511);
    h[i] = (c < 256) ? x[rr*256 + c] : m[rr*256 + (c-256)];
}
__global__ void k_gelu(float* p, long n){
    long i = blockIdx.x*(long)blockDim.x + threadIdx.x; if (i >= n) return;
    float x = p[i];
    p[i] = 0.5f * x * (1.f + erff(x * 0.70710678118654752f));
}
__global__ void k_add(float* o, const float* x, const float* m, long n){
    long i = blockIdx.x*(long)blockDim.x + threadIdx.x; if (i < n) o[i] = x[i] + m[i];
}
__global__ void k_catfeat(const float* f0, const float* f1, float* fb){
    long i = blockIdx.x*(long)blockDim.x + threadIdx.x;
    if (i >= (long)NB*LTOT*256) return;
    long b = i / ((long)LTOT*256), r = i % ((long)LTOT*256);
    long l = r >> 8; int c = (int)(r & 255);
    fb[i] = (l < LL) ? f0[(b*LL + l)*256 + c] : f1[(b*LL + (l-LL))*256 + c];
}
__global__ void k_bcast(const float* st, float* seeds){
    int i = blockIdx.x*blockDim.x + threadIdx.x;
    if (i < NB*NTOP*256) seeds[i] = st[i % (NTOP*256)];
}
__global__ void k_softmax_onehot(float* dm, float* top0, float* top1){
    long w = blockIdx.x*8L + (threadIdx.x >> 5);
    int lane = threadIdx.x & 31;
    if (w >= (long)NB*LTOT) return;
    int b = (int)(w / LTOT), m = (int)(w % LTOT);
    float* row = dm + w*NTOP;
    float mx = -1e30f;
    for (int c = lane; c < NTOP; c += 32) mx = fmaxf(mx, row[c]);
    mx = wave_max(mx);
    int am = 0x7fffffff;
    for (int c = lane; c < NTOP; c += 32) if (row[c] == mx && c < am) am = c;
    am = wave_imin(am);
    float sum = 0.f;
    for (int c = lane; c < NTOP; c += 32) sum += __expf(row[c]-mx);
    sum = wave_sum(sum);
    float inv = 1.f / sum;
    float* tp = (m < LL) ? (top0 + ((long)(b*LL + m))*NTOP)
                         : (top1 + ((long)(b*LL + (m-LL)))*NTOP);
    for (int c = lane; c < NTOP; c += 32){
        row[c] = __expf(row[c]-mx) * inv;
        tp[c] = (c == am) ? 1.f : 0.f;
    }
}
__global__ void k_theta(const float* dm, float* th0, float* th1){
    int half = blockIdx.x, b = blockIdx.y, k = threadIdx.x;
    if (k >= NTOP) return;
    const float* base = dm + ((long)b*LTOT + half*LL)*NTOP;
    float acc = 0.f;
    for (int m = 0; m < LL; ++m) acc += base[(long)m*NTOP + k];
    (half ? th1 : th0)[b*NTOP + k] = acc;
}
__global__ void k_topk(const float* th0, const float* th1, int* inds){
    int b = blockIdx.x;
    if (threadIdx.x) return;
    float s0 = 0.f, s1 = 0.f, th[NTOP];
    for (int k=0;k<NTOP;k++){ s0 += th0[b*NTOP+k]; s1 += th1[b*NTOP+k]; }
    s0 = fmaxf(s0, 1e-12f); s1 = fmaxf(s1, 1e-12f);
    float ts = 0.f;
    for (int k=0;k<NTOP;k++){ th[k] = (th0[b*NTOP+k]/s0)*(th1[b*NTOP+k]/s1); ts += th[k]; }
    ts = fmaxf(ts, 1e-12f);
    for (int k=0;k<NTOP;k++) th[k] /= ts;
    for (int kk=0;kk<NSAMP;kk++){
        int best = 0; float bv = -1e30f;
        for (int k=0;k<NTOP;k++) if (th[k] > bv){ bv = th[k]; best = k; }
        inds[b*NSAMP + kk] = best; th[best] = -1e30f;
    }
}
__global__ void k_make_mask(const float* topics, const int* inds, int kk, float* t, float* tsum){
    int i = blockIdx.x*blockDim.x + threadIdx.x;
    if (i >= NB*LL) return;
    int b = i / LL, l = i % LL;
    float v = topics[((long)(b*LL + l))*NTOP + inds[b*NSAMP + kk]];
    t[i] = v; tsum[i] += v;
}
__global__ void k_valid(const float* t0, const float* t1, float* validf){
    __shared__ float a[1024], c[1024];
    float x0 = 0.f, x1 = 0.f;
    for (int i = threadIdx.x; i < NB*LL; i += 1024){ x0 += t0[i]; x1 += t1[i]; }
    a[threadIdx.x] = x0; c[threadIdx.x] = x1; __syncthreads();
    for (int s = 512; s > 0; s >>= 1){
        if (threadIdx.x < s){ a[threadIdx.x] += a[threadIdx.x+s]; c[threadIdx.x] += c[threadIdx.x+s]; }
        __syncthreads();
    }
    if (!threadIdx.x) validf[0] = (a[0] > 0.f && c[0] > 0.f) ? 1.f : 0.f;
}
__global__ void k_accu(float* upd, const float* validf, const float* t, const float* f, long n){
    long i = blockIdx.x*(long)blockDim.x + threadIdx.x; if (i >= n) return;
    upd[i] += validf[0] * t[i >> 8] * f[i];
}
__global__ void k_fmix(float* o, const float* feat, const float* tsum, const float* upd, long n){
    long i = blockIdx.x*(long)blockDim.x + threadIdx.x; if (i >= n) return;
    o[i] = (1.f - tsum[i >> 8]) * feat[i] + upd[i];
}
__global__ void k_rowstats(const float* conf, float* rmax, float* rsum){
    long w = blockIdx.x*8L + (threadIdx.x >> 5);
    if (w >= (long)NB*LL) return;
    int lane = threadIdx.x & 31;
    const float* row = conf + w * LL;
    float m = -1e30f, s = 0.f;
    for (int c = lane; c < LL; c += 32){
        float x = row[c];
        if (x > m){ s = s*__expf(m - x) + 1.f; m = x; } else s += __expf(x - m);
    }
    for (int o = 16; o > 0; o >>= 1){
        float m2 = __shfl_xor(m, o, 32), s2 = __shfl_xor(s, o, 32);
        float nm = fmaxf(m, m2);
        s = s*__expf(m - nm) + s2*__expf(m2 - nm); m = nm;
    }
    if (!lane){ rmax[w] = m; rsum[w] = s; }
}
__global__ void k_colstats(const float* conf, float* cmax, float* csum){
    int b = blockIdx.y, sc = blockIdx.x*256 + threadIdx.x;
    if (sc >= LL) return;
    const float* base = conf + (long)b*LL*LL;
    float m = -1e30f, s = 0.f;
    for (int l = 0; l < LL; ++l){
        float x = base[(long)l*LL + sc];
        if (x > m){ s = s*__expf(m - x) + 1.f; m = x; } else s += __expf(x - m);
    }
    cmax[b*LL + sc] = m; csum[b*LL + sc] = s;
}
__global__ void k_conf_final(const float* raw, const float* rmax, const float* rsum,
                             const float* cmax, const float* csum, float* out){
    long i = blockIdx.x*(long)blockDim.x + threadIdx.x;
    if (i >= (long)NB*LL*LL) return;
    long b = i / ((long)LL*LL), r = i % ((long)LL*LL);
    int l = (int)(r / LL), sc = (int)(r % LL);
    float x = raw[i];
    float p1 = __expf(x - cmax[b*LL + sc]) / csum[b*LL + sc]; // softmax axis=1
    float p2 = __expf(x - rmax[b*LL + l]) / rsum[b*LL + l];   // softmax axis=2
    out[i] = p1 * p2;
}

// ---------------- host orchestration ----------------
struct LayerP { const float *wq,*wk,*wv,*wm,*w1,*w2,*g1,*b1,*g2,*b2; };
struct LayerT { const float *wqT,*wkT,*wvT,*wmT,*w1T,*w2T,*g1,*b1,*g2,*b2; };
struct Scr { float *q,*k,*v,*KV,*Ksum,*msg,*t1b,*t2b,*h1,*h2; };

static inline unsigned cdiv(long a, long b){ return (unsigned)((a + b - 1)/b); }

static void launch_gemm(const float* A, const float* B, float* C,
                        int M, int N, int K, long sA, long sB, long sC,
                        float alpha, hipStream_t st)
{
    dim3 g(cdiv(N,64), cdiv(M,64), NB);
    k_gemm<<<g, 128, 0, st>>>(A, B, C, M, N, K, sA, sB, sC, alpha);
}

static void encoder(const LayerT& p, const float* x, int Lx, const float* src, int Ls,
                    const float* xm, const float* sm, float* out, const Scr& S, hipStream_t st)
{
    long rx = (long)NB*Lx, rs = (long)NB*Ls;
    long nx = rx*256, ns = rs*256;
    launch_gemm(x,   p.wqT, S.q, Lx, 256, 256, (long)Lx*256, 0, (long)Lx*256, 1.f, st);
    launch_gemm(src, p.wkT, S.k, Ls, 256, 256, (long)Ls*256, 0, (long)Ls*256, 1.f, st);
    launch_gemm(src, p.wvT, S.v, Ls, 256, 256, (long)Ls*256, 0, (long)Ls*256, 1.f, st);
    k_actmask<<<cdiv(nx,256),256,0,st>>>(S.q, xm, nx, 1);
    k_actmask<<<cdiv(ns,256),256,0,st>>>(S.k, sm, ns, 1);
    if (sm) k_actmask<<<cdiv(ns,256),256,0,st>>>(S.v, sm, ns, 0);
    k_kv<<<NB*NHEAD, 1024, 0, st>>>(S.k, S.v, S.KV, S.Ksum, Ls);
    k_attn_out<<<cdiv(rx*NHEAD,8),256,0,st>>>(S.q, S.KV, S.Ksum, S.msg, Lx);
    launch_gemm(S.msg, p.wmT, S.t1b, Lx, 256, 256, (long)Lx*256, 0, (long)Lx*256, 1.f, st);
    k_ln<<<cdiv(rx,8),256,0,st>>>(S.t1b, p.g1, p.b1, S.t1b, rx);
    k_concat<<<cdiv(rx*512,256),256,0,st>>>(x, S.t1b, S.h1, rx);
    launch_gemm(S.h1, p.w1T, S.h2, Lx, 512, 512, (long)Lx*512, 0, (long)Lx*512, 1.f, st);
    k_gelu<<<cdiv(rx*512,256),256,0,st>>>(S.h2, rx*512);
    launch_gemm(S.h2, p.w2T, S.t2b, Lx, 256, 512, (long)Lx*512, 0, (long)Lx*256, 1.f, st);
    k_ln<<<cdiv(rx,8),256,0,st>>>(S.t2b, p.g2, p.b2, S.t2b, rx);
    k_add<<<cdiv(nx,256),256,0,st>>>(out, x, S.t2b, nx);
}

static LayerP getLayer(void* const* d_in, int base){
    // pytree leaf order: dict keys sorted: b1,b2,g1,g2,w1,w2,wk,wm,wq,wv
    LayerP p;
    p.b1 = (const float*)d_in[base+0]; p.b2 = (const float*)d_in[base+1];
    p.g1 = (const float*)d_in[base+2]; p.g2 = (const float*)d_in[base+3];
    p.w1 = (const float*)d_in[base+4]; p.w2 = (const float*)d_in[base+5];
    p.wk = (const float*)d_in[base+6]; p.wm = (const float*)d_in[base+7];
    p.wq = (const float*)d_in[base+8]; p.wv = (const float*)d_in[base+9];
    return p;
}

extern "C" void kernel_launch(void* const* d_in, const int* in_sizes, int n_in,
                              void* d_out, int out_size, void* d_ws, size_t ws_size,
                              hipStream_t stream)
{
    (void)in_sizes; (void)n_in; (void)out_size; (void)ws_size;
    hipStream_t st = stream;
    const float* feat0_in = (const float*)d_in[0];
    const float* feat1_in = (const float*)d_in[1];
    LayerP Lp[8];
    for (int i = 0; i < 4; ++i) Lp[i]   = getLayer(d_in, 2 + 10*i);
    const float* seed_tokens = (const float*)d_in[42];
    for (int i = 0; i < 4; ++i) Lp[4+i] = getLayer(d_in, 43 + 10*i);

    float* o_feat0 = (float*)d_out;
    float* o_feat1 = o_feat0 + (long)NB*LL*256;
    float* o_conf  = o_feat1 + (long)NB*LL*256;
    float* o_top0  = o_conf  + (long)NB*LL*LL;
    float* o_top1  = o_top0  + (long)NB*LL*NTOP;

    // ---- workspace carve (deterministic bump allocator) ----
    char* wp = (char*)d_ws;
    auto allocf = [&](long nf)->float*{
        float* p = (float*)wp; wp += ((nf*4 + 255)/256)*256; return p; };
    const long NF = (long)NB*LL*256;
    const long NS = (long)NB*LTOT*256;
    // transposed weights: all GEMMs take the [N,K] (NT) fast path
    LayerT Lt[8];
    for (int i = 0; i < 8; ++i) {
        float* wq = allocf(256*256); float* wk = allocf(256*256);
        float* wv = allocf(256*256); float* wm = allocf(256*256);
        float* w1 = allocf(512*512); float* w2 = allocf(256*512);
        Lt[i].wqT=wq; Lt[i].wkT=wk; Lt[i].wvT=wv; Lt[i].wmT=wm; Lt[i].w1T=w1; Lt[i].w2T=w2;
        Lt[i].g1=Lp[i].g1; Lt[i].b1=Lp[i].b1; Lt[i].g2=Lp[i].g2; Lt[i].b2=Lp[i].b2;
        k_transpose<<<cdiv(256*256,256),256,0,st>>>(Lp[i].wq, wq, 256, 256);
        k_transpose<<<cdiv(256*256,256),256,0,st>>>(Lp[i].wk, wk, 256, 256);
        k_transpose<<<cdiv(256*256,256),256,0,st>>>(Lp[i].wv, wv, 256, 256);
        k_transpose<<<cdiv(256*256,256),256,0,st>>>(Lp[i].wm, wm, 256, 256);
        k_transpose<<<cdiv(512*512,256),256,0,st>>>(Lp[i].w1, w1, 512, 512);
        k_transpose<<<cdiv(512*256,256),256,0,st>>>(Lp[i].w2, w2, 512, 256);
    }
    float* featbuf = allocf(NS);
    float* seedsA  = allocf(NB*NTOP*256);
    float* seedsB  = allocf(NB*NTOP*256);
    float* f0a = allocf(NF); float* f0b = allocf(NF);
    float* f1a = allocf(NF); float* f1b = allocf(NF);
    float* tf0a = allocf(NF); float* tf0b = allocf(NF);
    float* tf1a = allocf(NF); float* tf1b = allocf(NF);
    float* upd0 = allocf(NF); float* upd1 = allocf(NF);
    float* tsum0 = allocf(NB*LL); float* tsum1 = allocf(NB*LL);
    float* t0m = allocf(NB*LL);   float* t1m = allocf(NB*LL);
    float* validf = allocf(64);
    int*   inds = (int*)allocf(64);
    float* KV = allocf(NB*NHEAD*1024); float* Ksum = allocf(NB*NHEAD*32);
    float* dm = allocf((long)NB*LTOT*NTOP);
    float* th0 = allocf(256); float* th1 = allocf(256);
    float* rmax = allocf(NB*LL); float* rsum = allocf(NB*LL);
    float* cmax = allocf(NB*LL); float* csum = allocf(NB*LL);
    // union region: encoder scratch (~118MB) overlaps conf raw (~184MB)
    char* uni = wp;
    Scr S;
    {
        char* up = uni;
        auto af = [&](long nf)->float*{ float* p=(float*)up; up += ((nf*4+255)/256)*256; return p; };
        S.q = af(NF); S.k = af(NS); S.v = af(NS);
        S.msg = af(NF); S.t1b = af(NF); S.t2b = af(NF);
        S.h1 = af((long)NB*LL*512); S.h2 = af((long)NB*LL*512);
    }
    float* confraw = (float*)uni;
    S.KV = KV; S.Ksum = Ksum;

    // ---- init accumulators ----
    k_zero<<<cdiv(NF,256),256,0,st>>>(upd0, NF);
    k_zero<<<cdiv(NF,256),256,0,st>>>(upd1, NF);
    k_zero<<<cdiv(NB*LL,256),256,0,st>>>(tsum0, NB*LL);
    k_zero<<<cdiv(NB*LL,256),256,0,st>>>(tsum1, NB*LL);

    // ---- main stack ----
    k_catfeat<<<cdiv(NS,256),256,0,st>>>(feat0_in, feat1_in, featbuf);
    k_bcast<<<cdiv(NB*NTOP*256,256),256,0,st>>>(seed_tokens, seedsA);
    encoder(Lt[0], seedsA, NTOP, featbuf, LTOT, nullptr, nullptr, seedsB, S, st);
    encoder(Lt[1], feat0_in, LL, seedsB, NTOP, nullptr, nullptr, f0a, S, st);
    encoder(Lt[1], feat1_in, LL, seedsB, NTOP, nullptr, nullptr, f1a, S, st);
    encoder(Lt[2], seedsB, NTOP, featbuf, LTOT, nullptr, nullptr, seedsA, S, st);
    encoder(Lt[3], f0a, LL, seedsA, NTOP, nullptr, nullptr, f0b, S, st);
    encoder(Lt[3], f1a, LL, seedsA, NTOP, nullptr, nullptr, f1b, S, st);
    float* fe0 = f0b; float* fe1 = f1b; float* seeds = seedsA;

    // ---- topic assignment (stale concat feat x final seeds) ----
    launch_gemm(featbuf, seeds, dm, LTOT, NTOP, 256,
                (long)LTOT*256, (long)NTOP*256, (long)LTOT*NTOP, 1.f, st);
    k_softmax_onehot<<<cdiv((long)NB*LTOT,8),256,0,st>>>(dm, o_top0, o_top1);
    k_theta<<<dim3(2,NB),128,0,st>>>(dm, th0, th1);
    k_topk<<<NB,32,0,st>>>(th0, th1, inds);

    // ---- topic refinement loop ----
    for (int kk = 0; kk < NSAMP; ++kk) {
        k_make_mask<<<cdiv(NB*LL,256),256,0,st>>>(o_top0, inds, kk, t0m, tsum0);
        k_make_mask<<<cdiv(NB*LL,256),256,0,st>>>(o_top1, inds, kk, t1m, tsum1);
        k_valid<<<1,1024,0,st>>>(t0m, t1m, validf);
        k_copy<<<cdiv(NF,256),256,0,st>>>(tf0a, fe0, NF);
        k_copy<<<cdiv(NF,256),256,0,st>>>(tf1a, fe1, NF);
        float *c0 = tf0a, *n0 = tf0b, *c1 = tf1a, *n1 = tf1b, *tmp;
        for (int it = 0; it < NITER; ++it) {
            encoder(Lt[4+2*it], c0, LL, c0, LL, t0m, t0m, n0, S, st); tmp=c0;c0=n0;n0=tmp;
            encoder(Lt[4+2*it], c1, LL, c1, LL, t1m, t1m, n1, S, st); tmp=c1;c1=n1;n1=tmp;
            encoder(Lt[4+2*it+1], c0, LL, c1, LL, t0m, t1m, n0, S, st); tmp=c0;c0=n0;n0=tmp;
            encoder(Lt[4+2*it+1], c1, LL, c0, LL, t1m, t0m, n1, S, st); tmp=c1;c1=n1;n1=tmp;
        }
        k_accu<<<cdiv(NF,256),256,0,st>>>(upd0, validf, t0m, c0, NF);
        k_accu<<<cdiv(NF,256),256,0,st>>>(upd1, validf, t1m, c1, NF);
    }
    k_fmix<<<cdiv(NF,256),256,0,st>>>(o_feat0, fe0, tsum0, upd0, NF);
    k_fmix<<<cdiv(NF,256),256,0,st>>>(o_feat1, fe1, tsum1, upd1, NF);

    // ---- conf = softmax_axis1(raw) * softmax_axis2(raw), raw = feat0 feat1^T / 16 ----
    launch_gemm(o_feat0, o_feat1, confraw, LL, LL, 256,
                (long)LL*256, (long)LL*256, (long)LL*LL, 1.f/16.f, st);
    k_rowstats<<<cdiv((long)NB*LL,8),256,0,st>>>(confraw, rmax, rsum);
    k_colstats<<<dim3(cdiv(LL,256),NB),256,0,st>>>(confraw, cmax, csum);
    k_conf_final<<<cdiv((long)NB*LL*LL,256),256,0,st>>>(confraw, rmax, rsum, cmax, csum, o_conf);
}
